// AutoregressiveLSTM_19172734009351
// MI455X (gfx1250) — matrix operand order repaired and checked by reading, not compiled, not run on hardware
//
#include <hip/hip_runtime.h>
#include <hip/hip_bf16.h>

// ---------------------------------------------------------------------------
// AutoregressiveLSTM on MI455X (gfx1250):
//   per step: z = x*kernel + h @ R + bias  (h@R via v_wmma_f32_16x16x32_bf16)
//   gates fused in epilogue, c kept fp32, h stored bf16 row-major,
//   R pre-packed once into WMMA B-fragment order (L2-resident, 8 MB).
// ---------------------------------------------------------------------------

#define UNITS 1024
#define BROWS 4096
#define STEPS 128

typedef __attribute__((ext_vector_type(16))) __bf16      v16bf;
typedef __attribute__((ext_vector_type(8)))  float       v8f;
typedef __attribute__((ext_vector_type(4)))  unsigned int u32x4;

union Frag { u32x4 u[2]; v16bf h; };

static __device__ __forceinline__ unsigned short f2bf(float f) {
    unsigned int u = __float_as_uint(f);
    u += 0x7FFFu + ((u >> 16) & 1u);          // round-to-nearest-even
    return (unsigned short)(u >> 16);
}
static __device__ __forceinline__ float bf2f(unsigned short s) {
    return __uint_as_float(((unsigned int)s) << 16);
}
static __device__ __forceinline__ float sigm(float x) {
    return 1.0f / (1.0f + __expf(-x));
}
static __device__ __forceinline__ float tanh_fast(float x) {
    return 2.0f / (1.0f + __expf(-2.0f * x)) - 1.0f;
}

// --- Pack recurrent_kernel [K=1024, N=4096] fp32 -> bf16 B-fragment order ---
// Fragment (n_tile, k_tile): 32 lanes x 16 halves, lane-major contiguous.
// B-layout (ISA 7.12.2): lane l -> column n = l&15, K = 8*(l>>4)+i (i<8), 16+8*(l>>4)+(i-8) (i>=8).
__global__ __launch_bounds__(256) void pack_rk(const float* __restrict__ R,
                                               unsigned short* __restrict__ Rp) {
    int tid   = blockIdx.x * 256 + threadIdx.x;     // 2,097,152 threads
    int ipair = tid & 7;
    int lane  = (tid >> 3) & 31;
    int kt    = (tid >> 8) & 31;
    int nt    = tid >> 13;                           // 0..255
    int n  = nt * 16 + (lane & 15);
    int k0 = kt * 32 + ((lane >> 4) << 3) + ipair * 2 + ((ipair >> 2) << 3);
    float f0 = R[(size_t)k0 * 4096 + n];
    float f1 = R[((size_t)k0 + 1) * 4096 + n];
    unsigned int w = (unsigned int)f2bf(f0) | ((unsigned int)f2bf(f1) << 16);
    ((unsigned int*)Rp)[(size_t)(nt * 32 + kt) * 256 + lane * 8 + ipair] = w;
}

// --- h0 = c0 = concat([features, features]) ; h bf16, c fp32 ----------------
__global__ __launch_bounds__(256) void init_state(const float* __restrict__ feat,
                                                  unsigned short* __restrict__ h,
                                                  float* __restrict__ c) {
    int tid = blockIdx.x * 256 + threadIdx.x;       // 4M threads
    int u = tid & (UNITS - 1);
    int b = tid >> 10;
    float v = feat[(size_t)b * (UNITS / 2) + (u & (UNITS / 2 - 1))];
    h[tid] = f2bf(v);
    c[tid] = v;
}

// --- One LSTM step: 8 waves/block, wave = 16Mx16N unit-tile, 4 gate accs ----
__global__ __launch_bounds__(256) void lstm_step(
    const unsigned short* __restrict__ hcur, unsigned short* __restrict__ hnxt,
    float* __restrict__ c, const unsigned short* __restrict__ Rp,
    const float* __restrict__ Wk, const float* __restrict__ bias,
    const float* __restrict__ preds, int t) {

    const int lane = threadIdx.x & 31;
    const int wave = threadIdx.x >> 5;
    const int mt   = blockIdx.y * 2 + (wave >> 2);   // 0..255 (M tiles)
    const int ntu  = blockIdx.x * 4 + (wave & 3);    // 0..63  (unit-N tiles)
    const int nl   = lane & 15;
    const int m_base = mt * 16;
    const int n      = ntu * 16 + nl;

    v8f acc0 = {}, acc1 = {}, acc2 = {}, acc3 = {};

    // A-fragment addressing: lane -> row (l&15), K chunks {kb..kb+7, kb+16..kb+23}
    const size_t hrow = (size_t)(m_base + nl) * UNITS;
    const int    akb  = (lane >> 4) << 3;            // 0 or 8
    const unsigned short* bbase = Rp + (size_t)lane * 16;

    #pragma unroll 4
    for (int kt = 0; kt < 32; ++kt) {
        const unsigned short* ap = hcur + hrow + kt * 32 + akb;
        Frag a;
        a.u[0] = *(const u32x4*)(ap);
        a.u[1] = *(const u32x4*)(ap + 16);
        __builtin_prefetch(ap + 32, 0, 3);           // -> global_prefetch_b8

        Frag b0, b1, b2, b3;
        {
            const unsigned short* bp = bbase + (size_t)((0 * 64 + ntu) * 32 + kt) * 512;
            b0.u[0] = *(const u32x4*)(bp);  b0.u[1] = *(const u32x4*)(bp + 8);
        }
        {
            const unsigned short* bp = bbase + (size_t)((1 * 64 + ntu) * 32 + kt) * 512;
            b1.u[0] = *(const u32x4*)(bp);  b1.u[1] = *(const u32x4*)(bp + 8);
        }
        {
            const unsigned short* bp = bbase + (size_t)((2 * 64 + ntu) * 32 + kt) * 512;
            b2.u[0] = *(const u32x4*)(bp);  b2.u[1] = *(const u32x4*)(bp + 8);
        }
        {
            const unsigned short* bp = bbase + (size_t)((3 * 64 + ntu) * 32 + kt) * 512;
            b3.u[0] = *(const u32x4*)(bp);  b3.u[1] = *(const u32x4*)(bp + 8);
        }

        acc0 = __builtin_amdgcn_wmma_f32_16x16x32_bf16(false, a.h, false, b0.h,
                                                       (short)0, acc0, false, false);
        acc1 = __builtin_amdgcn_wmma_f32_16x16x32_bf16(false, a.h, false, b1.h,
                                                       (short)0, acc1, false, false);
        acc2 = __builtin_amdgcn_wmma_f32_16x16x32_bf16(false, a.h, false, b2.h,
                                                       (short)0, acc2, false, false);
        acc3 = __builtin_amdgcn_wmma_f32_16x16x32_bf16(false, a.h, false, b3.h,
                                                       (short)0, acc3, false, false);
    }

    // Epilogue: C layout -> lane l, vgpr v: m = m_base + v + 8*(l>=16), n = l&15
    const int hi8 = (lane >> 4) << 3;
    const float w0 = Wk[n],            w1 = Wk[n + 1024];
    const float w2 = Wk[n + 2048],     w3 = Wk[n + 3072];
    const float c0b = bias[n],         c1b = bias[n + 1024];
    const float c2b = bias[n + 2048],  c3b = bias[n + 3072];

    #pragma unroll
    for (int v = 0; v < 8; ++v) {
        const int m = m_base + v + hi8;
        const float xm = (t > 0) ? preds[(size_t)m * STEPS + (t - 1)] : 0.0f;
        const float zi = acc0[v] + xm * w0 + c0b;
        const float zf = acc1[v] + xm * w1 + c1b;
        const float zg = acc2[v] + xm * w2 + c2b;
        const float zo = acc3[v] + xm * w3 + c3b;
        const float ig = sigm(zi);
        const float fg = sigm(zf);
        const float gg = tanh_fast(zg);
        const float og = sigm(zo);
        const size_t idx = (size_t)m * UNITS + n;
        const float cn = fg * c[idx] + ig * gg;
        c[idx] = cn;
        hnxt[idx] = f2bf(og * tanh_fast(cn));
    }
}

// --- pred[b] = h_new[b,:] @ dense_w + dense_b ; one wave per row ------------
__global__ __launch_bounds__(256) void dense_step(const unsigned short* __restrict__ h,
                                                  const float* __restrict__ dw,
                                                  const float* __restrict__ db,
                                                  float* __restrict__ out, int t) {
    const int lane = threadIdx.x & 31;
    const int row  = blockIdx.x * 8 + (threadIdx.x >> 5);
    const unsigned short* hp = h + (size_t)row * UNITS;
    float s = 0.0f;
    #pragma unroll 4
    for (int j = lane * 2; j < UNITS; j += 64) {
        unsigned int w = *(const unsigned int*)(hp + j);
        s += bf2f((unsigned short)(w & 0xFFFFu)) * dw[j];
        s += bf2f((unsigned short)(w >> 16)) * dw[j + 1];
    }
    #pragma unroll
    for (int off = 16; off > 0; off >>= 1) s += __shfl_xor(s, off, 32);
    if (lane == 0) out[(size_t)row * STEPS + t] = s + db[0];
}

extern "C" void kernel_launch(void* const* d_in, const int* in_sizes, int n_in,
                              void* d_out, int out_size, void* d_ws, size_t ws_size,
                              hipStream_t stream) {
    const float* feat = (const float*)d_in[0];   // [4096, 512]
    const float* Wk   = (const float*)d_in[1];   // [1, 4096]
    const float* R    = (const float*)d_in[2];   // [1024, 4096]
    const float* bias = (const float*)d_in[3];   // [4096]
    const float* dw   = (const float*)d_in[4];   // [1024]
    const float* db   = (const float*)d_in[5];   // [1]
    float* out = (float*)d_out;                  // [4096, 128, 1]

    char* ws = (char*)d_ws;
    unsigned short* Rp = (unsigned short*)(ws);                       //  8 MB
    unsigned short* hA = (unsigned short*)(ws + (size_t)(8  << 20));  //  8 MB
    unsigned short* hB = (unsigned short*)(ws + (size_t)(16 << 20));  //  8 MB
    float*          cS = (float*)        (ws + (size_t)(24 << 20));   // 16 MB

    pack_rk<<<8192, 256, 0, stream>>>(R, Rp);
    init_state<<<16384, 256, 0, stream>>>(feat, hA, cS);

    unsigned short* hc = hA;
    unsigned short* hn = hB;
    for (int t = 0; t < STEPS; ++t) {
        dim3 grid(16, 128);   // 64 unit-N tiles / 4 per block ; 256 M tiles / 2
        lstm_step<<<grid, 256, 0, stream>>>(hc, hn, cS, Rp, Wk, bias, out, t);
        dense_step<<<512, 256, 0, stream>>>(hn, dw, db, out, t);
        unsigned short* tmp = hc; hc = hn; hn = tmp;
    }
}
